// FUNASR_NANO_DECODER_MAIN_19688130085125
// MI455X (gfx1250) — matrix operand order repaired and checked by reading, not compiled, not run on hardware
//
#include <hip/hip_runtime.h>
#include <math.h>
#include <stdint.h>

typedef __attribute__((ext_vector_type(16))) _Float16 v16h;
typedef __attribute__((ext_vector_type(8)))  _Float16 v8h;
typedef __attribute__((ext_vector_type(8)))  float    v8f;
typedef __attribute__((ext_vector_type(4)))  int      v4i;

namespace {
constexpr int Ln = 4, Bn = 2, Sn = 1024, Dn = 1024;
constexpr int Hn = 16, KVn = 8, HDn = 128, Gn = Hn / KVn;
constexpr int Fn = 3072, Vn = 32000;
constexpr float EPSn = 1e-6f;
constexpr float SCALEn = 0.29730177875068026f;   // HD^-0.25 = 128^-0.25
}

// ---------------------------------------------------------------------------
// CDNA5 async global->LDS copy (16B per lane), guarded so compile never breaks.
// ---------------------------------------------------------------------------
#if __has_builtin(__builtin_amdgcn_global_load_async_to_lds_b128) && \
    __has_builtin(__builtin_amdgcn_s_wait_asynccnt)
#define USE_ASYNC_LDS 1
#else
#define USE_ASYNC_LDS 0
#endif

__device__ __forceinline__ void copy16_to_lds(const _Float16* g, _Float16* l) {
#if USE_ASYNC_LDS
  __builtin_amdgcn_global_load_async_to_lds_b128(
      (v4i*)(uintptr_t)g, (v4i*)(uintptr_t)l, 0, 0);
#else
  *reinterpret_cast<v8h*>(l) = *reinterpret_cast<const v8h*>(g);
#endif
}
__device__ __forceinline__ void async_wait_keep4() {
#if USE_ASYNC_LDS
  __builtin_amdgcn_s_wait_asynccnt(4);   // current tile done, next tile in flight
#endif
}
__device__ __forceinline__ void async_wait_all() {
#if USE_ASYNC_LDS
  __builtin_amdgcn_s_wait_asynccnt(0);
#endif
}

// ---------------------------------------------------------------------------
// WMMA helper: D = A(16x32 f16) * B(32x16 f16) + C(16x16 f32)
// ---------------------------------------------------------------------------
__device__ __forceinline__ v8f wmma_f16(v16h a, v16h b, v8f c) {
  return __builtin_amdgcn_wmma_f32_16x16x32_f16(
      /*neg_a=*/false, a, /*neg_b=*/false, b,
      /*c_mod=*/(short)0, c, /*reuse_a=*/false, /*reuse_b=*/false);
}

__device__ __forceinline__ v16h cat16(v8h lo, v8h hi) {
  return __builtin_shufflevector(lo, hi,
      0, 1, 2, 3, 4, 5, 6, 7, 8, 9, 10, 11, 12, 13, 14, 15);
}

// ---------------------------------------------------------------------------
// RMSNorm -> f16: one block per row.  out = (f16)(w * x * rsqrt(mean(x^2)+eps))
// ---------------------------------------------------------------------------
__global__ __launch_bounds__(256) void rmsnorm_kernel(
    const float* __restrict__ in, const float* __restrict__ w,
    _Float16* __restrict__ out, int Dd) {
  const size_t rb = (size_t)blockIdx.x * Dd;
  float ss = 0.f;
  for (int d = threadIdx.x; d < Dd; d += 256) {
    float x = in[rb + d];
    ss += x * x;
  }
#pragma unroll
  for (int off = 16; off >= 1; off >>= 1) ss += __shfl_xor(ss, off, 32);
  __shared__ float red[8];
  if ((threadIdx.x & 31) == 0) red[threadIdx.x >> 5] = ss;
  __syncthreads();
  float tot = red[0] + red[1] + red[2] + red[3] + red[4] + red[5] + red[6] + red[7];
  float inv = rsqrtf(tot / (float)Dd + EPSn);
  for (int d = threadIdx.x; d < Dd; d += 256)
    out[rb + d] = (_Float16)(w[d] * in[rb + d] * inv);
}

// ---------------------------------------------------------------------------
// Weight convert + transpose: W[K][N] fp32 -> Wt[N][K] f16 (32x32 LDS tiles).
// grid (K/32, N/32), 256 threads.
// ---------------------------------------------------------------------------
__global__ __launch_bounds__(256) void wtrans_kernel(
    const float* __restrict__ W, _Float16* __restrict__ Wt, int N, int K) {
  __shared__ _Float16 tile[32][33];
  const int k0 = blockIdx.x * 32;
  const int n0 = blockIdx.y * 32;
  const int ls = threadIdx.x & 31, lr = threadIdx.x >> 5;
#pragma unroll
  for (int i = 0; i < 4; ++i) {
    const int k = k0 + lr + i * 8;
    tile[lr + i * 8][ls] = (_Float16)W[(size_t)k * N + n0 + ls];
  }
  __syncthreads();
#pragma unroll
  for (int i = 0; i < 4; ++i) {
    const int n = n0 + lr + i * 8;
    Wt[(size_t)n * K + k0 + ls] = tile[ls][lr + i * 8];
  }
}

// ---------------------------------------------------------------------------
// GEMM: C[M,N] = A[M,K] @ Wt[N,K]^T (+ Res optional).
// A f16 row-major, Wt f16 row-major (pre-transposed weights), C/Res fp32.
// 128x128 block tile, BK=32, double-buffered LDS filled by async 16B copies.
// Block = 256 threads = 8 waves arranged 4(M) x 2(N); wave strip = 32x64
// (2 A fragments x 4 B fragments = 8 WMMAs per K-step per wave).
// ---------------------------------------------------------------------------
__global__ __launch_bounds__(256) void gemm_wmma_kernel(
    const _Float16* __restrict__ A, const _Float16* __restrict__ Wt,
    const float* __restrict__ Res, float* __restrict__ C,
    int M, int N, int K) {
  constexpr int BM = 128, BN = 128, BK = 32;
  constexpr int AST = 40;    // A row stride in f16 (80B, 16B-aligned)
  constexpr int BKP = 40;    // B col stride in f16 (80B)
  __shared__ _Float16 Ash[2][BM * AST];   // [row][k]
  __shared__ _Float16 Bsh[2][BN * BKP];   // [col][k]

  const int nBlocks = N / BN;
  const int mBase = (blockIdx.x / nBlocks) * BM;
  const int nBase = (blockIdx.x % nBlocks) * BN;
  if (mBase >= M) return;

  const int tid  = threadIdx.x;
  const int lane = tid & 31;
  const int wv   = tid >> 5;
  const int waveM = wv >> 1;        // 0..3 -> 32-row strip
  const int waveN = wv & 1;         // 0..1 -> 64-col strip
  const int half  = lane >> 4;      // K-half select per ISA fragment layout
  const int l15   = lane & 15;

  // staging coordinates: A 2x16B per thread, B 2x16B per thread
  const int ar = tid >> 1;                     // A row 0..127
  const int aseg = (tid & 1) * 16;             // A k-offset 0 or 16 (f16)
  const int bcol = tid >> 1;                   // B column 0..127
  const int bhalf = (tid & 1) * 16;            // B k-offset 0 or 16 (f16)
  const _Float16* Ag = A + (size_t)(mBase + ar) * K + aseg;
  const _Float16* Wg = Wt + (size_t)(nBase + bcol) * K + bhalf;
  _Float16* AshA = &Ash[0][0];
  _Float16* BshA = &Bsh[0][0];
  const int aoff = ar * AST + aseg;
  const int boff = bcol * BKP + bhalf;

  auto issue = [&](int k0, int buf) {
    copy16_to_lds(Ag + k0,     AshA + buf * (BM * AST) + aoff);
    copy16_to_lds(Ag + k0 + 8, AshA + buf * (BM * AST) + aoff + 8);
    copy16_to_lds(Wg + k0,     BshA + buf * (BN * BKP) + boff);
    copy16_to_lds(Wg + k0 + 8, BshA + buf * (BN * BKP) + boff + 8);
  };

  v8f acc[2][4];
#pragma unroll
  for (int m = 0; m < 2; ++m)
#pragma unroll
    for (int s = 0; s < 4; ++s) acc[m][s] = v8f{};

  const int T = K / BK;
  issue(0, 0);

  for (int i = 0; i < T; ++i) {
    const int cur = i & 1;
    if (i + 1 < T) {
      issue((i + 1) * BK, cur ^ 1);
      async_wait_keep4();
    } else {
      async_wait_all();
    }
    __syncthreads();   // current tile visible to all waves

    // ---- two A fragments (rows waveM*32 .. +31), each two 16B LDS loads
    const _Float16* ab0 = &Ash[cur][(waveM * 32 + l15) * AST + half * 8];
    const _Float16* ab1 = ab0 + 16 * AST;
    v16h af0 = cat16(*reinterpret_cast<const v8h*>(ab0),
                     *reinterpret_cast<const v8h*>(ab0 + 16));
    v16h af1 = cat16(*reinterpret_cast<const v8h*>(ab1),
                     *reinterpret_cast<const v8h*>(ab1 + 16));

    // ---- 4 B fragments, each reused by both A fragments
    const int colb = waveN * 64 + l15;
#pragma unroll
    for (int s = 0; s < 4; ++s) {
      const _Float16* bb = &Bsh[cur][(colb + s * 16) * BKP + half * 8];
      v16h bf = cat16(*reinterpret_cast<const v8h*>(bb),
                      *reinterpret_cast<const v8h*>(bb + 16));
      acc[0][s] = wmma_f16(af0, bf, acc[0][s]);
      acc[1][s] = wmma_f16(af1, bf, acc[1][s]);
    }

    __syncthreads();   // done reading buf[cur]; it may be overwritten next iter
  }

  // ---- store (C layout: VGPR r -> row +r+8*half, col +l15 (+16 per acc))
#pragma unroll
  for (int m = 0; m < 2; ++m) {
    const int rowB = mBase + waveM * 32 + m * 16 + (half << 3);
    const int colB = nBase + waveN * 64 + l15;
#pragma unroll
    for (int r = 0; r < 8; ++r) {
      const size_t idx = (size_t)(rowB + r) * N + colB;
      float o0 = acc[m][0][r], o1 = acc[m][1][r];
      float o2 = acc[m][2][r], o3 = acc[m][3][r];
      if (Res) {
        o0 += Res[idx];      o1 += Res[idx + 16];
        o2 += Res[idx + 32]; o3 += Res[idx + 48];
      }
      C[idx]      = o0; C[idx + 16] = o1;
      C[idx + 32] = o2; C[idx + 48] = o3;
    }
  }
}

// ---------------------------------------------------------------------------
// Per-head RMSNorm + RoPE; emits f16 directly in attention-friendly layout:
// in : fp32 (B,S,nH,HD), out: f16 (B,nH,S,HD).
// ---------------------------------------------------------------------------
__global__ __launch_bounds__(HDn) void qknorm_rope_kernel(
    const float* __restrict__ t, const float* __restrict__ nw,
    _Float16* __restrict__ out16, int nH, int history) {
  const int blk = blockIdx.x;                 // (b*S + s)*nH + h
  const int h  = blk % nH;
  const int bs = blk / nH;
  const int s  = bs % Sn;
  const int b  = bs / Sn;
  const float* row = t + (size_t)blk * HDn;
  const int d = threadIdx.x;
  float v = row[d];

  __shared__ float sh[HDn];
  sh[d] = v * v;
  __syncthreads();
#pragma unroll
  for (int off = HDn / 2; off >= 1; off >>= 1) {
    if (d < off) sh[d] += sh[d + off];
    __syncthreads();
  }
  const float inv = rsqrtf(sh[0] / (float)HDn + EPSn);
  __syncthreads();

  __shared__ float shn[HDn];
  const float xn = v * inv * nw[d];
  shn[d] = xn;
  __syncthreads();

  const int dm = d & (HDn / 2 - 1);
  const float pos = (float)(history + s);
  const float invf = expf(-(float)(2 * dm) / (float)HDn * 9.210340371976184f); // ln(1e4)
  const float th = pos * invf;
  const float c  = cosf(th) * SCALEn;
  const float sn = sinf(th) * SCALEn;
  const float rot = (d < HDn / 2) ? -shn[d + HDn / 2] : shn[d - HDn / 2];
  out16[((size_t)((b * nH + h) * Sn + s)) * HDn + d] = (_Float16)(xn * c + rot * sn);
}

// ---------------------------------------------------------------------------
// V transpose + f16 convert: (B,S,KV,HD) fp32 -> (B,KV,HD,S) f16.
// ---------------------------------------------------------------------------
__global__ __launch_bounds__(256) void vtrans_kernel(
    const float* __restrict__ v, _Float16* __restrict__ vt) {
  __shared__ _Float16 tile[32][33];
  const int bk = blockIdx.x;
  const int s0 = blockIdx.y * 32;
  const int h0 = blockIdx.z * 32;
  const int b = bk / KVn, kvh = bk % KVn;
  const int ls = threadIdx.x & 31, lr = threadIdx.x >> 5;
#pragma unroll
  for (int i = 0; i < 4; ++i) {
    const int s = s0 + lr + i * 8;
    tile[lr + i * 8][ls] =
        (_Float16)v[((size_t)((b * Sn + s) * KVn + kvh)) * HDn + h0 + ls];
  }
  __syncthreads();
#pragma unroll
  for (int i = 0; i < 4; ++i) {
    const int hd = h0 + lr + i * 8;
    vt[((size_t)((b * KVn + kvh) * HDn + hd)) * Sn + s0 + ls] = tile[ls][lr + i * 8];
  }
}

// ---------------------------------------------------------------------------
// Flash attention (causal, additive -128*mask_scale mask as in reference).
// One wave per (b, h, 16-query-row tile).  qh (B,H,S,HD), kh (B,KV,S,HD),
// vt (B,KV,HD,S), all f16; ctx written as f16 (B,S,H,HD).
// ---------------------------------------------------------------------------
__global__ __launch_bounds__(32) void attn_kernel(
    const _Float16* __restrict__ qh, const _Float16* __restrict__ kh,
    const _Float16* __restrict__ vt, const float* __restrict__ mask_scale,
    _Float16* __restrict__ ctx) {
  int idx = blockIdx.x;
  const int mT = idx % (Sn / 16); idx /= (Sn / 16);
  const int h  = idx % Hn;
  const int b  = idx / Hn;
  const int kvh = h / Gn;
  const int m0 = mT * 16;
  const int lane = threadIdx.x;
  const int half = lane >> 4;
  const int l15  = lane & 15;
  const float mneg = -128.0f * mask_scale[0];

  __shared__ _Float16 pbuf[16 * 32];

  // ---- q as 4 A fragments covering hd 0..127
  v16h qa[4];
  const _Float16* qrow = qh + ((size_t)((b * Hn + h) * Sn + m0 + l15)) * HDn;
#pragma unroll
  for (int c = 0; c < 4; ++c) {
    const _Float16* qp = qrow + c * 32 + half * 8;
    qa[c] = cat16(*reinterpret_cast<const v8h*>(qp),
                  *reinterpret_cast<const v8h*>(qp + 16));
  }

  float mrow[8], lrow[8];
  v8f o[8];
#pragma unroll
  for (int r = 0; r < 8; ++r) { mrow[r] = -1e30f; lrow[r] = 0.f; o[r] = v8f{}; }

  const _Float16* kbaseP = kh + ((size_t)((b * KVn + kvh) * Sn)) * HDn;
  const _Float16* vbaseP = vt + ((size_t)((b * KVn + kvh) * HDn)) * Sn;

  for (int n0 = 0; n0 <= m0 + 15; n0 += 32) {
    // ---- scores: two 16x16 tiles (cols n0.. and n0+16..)
    v8f sc0 = {}, sc1 = {};
#pragma unroll
    for (int c = 0; c < 4; ++c) {
      const _Float16* kr0 = kbaseP + (size_t)(n0 + l15) * HDn + c * 32 + half * 8;
      const _Float16* kr1 = kr0 + (size_t)16 * HDn;
      v16h bk0 = cat16(*reinterpret_cast<const v8h*>(kr0),
                       *reinterpret_cast<const v8h*>(kr0 + 16));
      v16h bk1 = cat16(*reinterpret_cast<const v8h*>(kr1),
                       *reinterpret_cast<const v8h*>(kr1 + 16));
      sc0 = wmma_f16(qa[c], bk0, sc0);
      sc1 = wmma_f16(qa[c], bk1, sc1);
    }

    // ---- mask + online softmax (row = m0 + r + 8*half, replicated per half)
#pragma unroll
    for (int r = 0; r < 8; ++r) {
      const int grow = m0 + r + (half << 3);
      const int c0 = n0 + l15, c1 = c0 + 16;
      float s0v = sc0[r] + (c0 > grow ? mneg : 0.f);
      float s1v = sc1[r] + (c1 > grow ? mneg : 0.f);
      float mx = fmaxf(s0v, s1v);
#pragma unroll
      for (int off = 1; off < 16; off <<= 1) mx = fmaxf(mx, __shfl_xor(mx, off, 32));
      const float mnew  = fmaxf(mrow[r], mx);
      const float alpha = expf(mrow[r] - mnew);
      const float p0 = expf(s0v - mnew);
      const float p1 = expf(s1v - mnew);
      float ps = p0 + p1;
#pragma unroll
      for (int off = 1; off < 16; off <<= 1) ps += __shfl_xor(ps, off, 32);
      lrow[r] = lrow[r] * alpha + ps;
      mrow[r] = mnew;
#pragma unroll
      for (int j = 0; j < 8; ++j) o[j][r] = o[j][r] * alpha;
      const int rr = r + (half << 3);
      pbuf[rr * 32 + l15]      = (_Float16)p0;
      pbuf[rr * 32 + 16 + l15] = (_Float16)p1;
    }

    asm volatile("s_wait_dscnt 0x0" ::: "memory");   // LDS stores -> loads (1 wave)

    // ---- P (16x32) reloaded in A-fragment layout
    v16h pa;
    {
      const int cb = half * 8;
#pragma unroll
      for (int j = 0; j < 8; ++j) {
        pa[j]     = pbuf[l15 * 32 + cb + j];
        pa[8 + j] = pbuf[l15 * 32 + 16 + cb + j];
      }
    }

    // ---- ctx += P @ Vtile(32 x 128): 8 B fragments, each 2 x 16B loads
#pragma unroll
    for (int j = 0; j < 8; ++j) {
      const _Float16* vr = vbaseP + (size_t)(j * 16 + l15) * Sn + n0 + half * 8;
      v16h vb = cat16(*reinterpret_cast<const v8h*>(vr),
                      *reinterpret_cast<const v8h*>(vr + 16));
      o[j] = wmma_f16(pa, vb, o[j]);
    }
  }

  // ---- epilogue: ctx[b,s,h,:] = (f16)(o / l)
#pragma unroll
  for (int r = 0; r < 8; ++r) {
    const int grow = m0 + r + (half << 3);
    const float invl = 1.0f / lrow[r];
    _Float16* crow = ctx + ((size_t)((b * Sn + grow) * Hn + h)) * HDn;
#pragma unroll
    for (int j = 0; j < 8; ++j) crow[j * 16 + l15] = (_Float16)(o[j][r] * invl);
  }
}

// ---------------------------------------------------------------------------
// gh = (f16)(silu(g) * u)
// ---------------------------------------------------------------------------
__global__ __launch_bounds__(256) void silu_mul_kernel(
    const float* __restrict__ g, const float* __restrict__ u,
    _Float16* __restrict__ gh, size_t n) {
  const size_t i = (size_t)blockIdx.x * 256 + threadIdx.x;
  if (i < n) {
    const float x = g[i];
    gh[i] = (_Float16)((x / (1.f + expf(-x))) * u[i]);
  }
}

// ---------------------------------------------------------------------------
// last-token RMSNorm: xl[b,:] = rms(x[b, S-1, :], norm_w)  (fp32)
// ---------------------------------------------------------------------------
__global__ __launch_bounds__(256) void last_rms_kernel(
    const float* __restrict__ x, const float* __restrict__ w,
    float* __restrict__ xl) {
  const int b = blockIdx.x;
  const float* row = x + ((size_t)(b * Sn + (Sn - 1))) * Dn;
  float ss = 0.f;
  for (int d = threadIdx.x; d < Dn; d += 256) { float t = row[d]; ss += t * t; }
#pragma unroll
  for (int off = 16; off >= 1; off >>= 1) ss += __shfl_xor(ss, off, 32);
  __shared__ float red[8];
  if ((threadIdx.x & 31) == 0) red[threadIdx.x >> 5] = ss;
  __syncthreads();
  float tot = red[0] + red[1] + red[2] + red[3] + red[4] + red[5] + red[6] + red[7];
  float inv = rsqrtf(tot / (float)Dn + EPSn);
  for (int d = threadIdx.x; d < Dn; d += 256) xl[(size_t)b * Dn + d] = w[d] * row[d] * inv;
}

// ---------------------------------------------------------------------------
// LM head GEMV: out[b, v] = xl[b,:] . W[:, v]   (M=2, HBM-bound -> scalar fp32)
// ---------------------------------------------------------------------------
__global__ __launch_bounds__(256) void lm_head_kernel(
    const float* __restrict__ xl, const float* __restrict__ W,
    float* __restrict__ out) {
  const int b = blockIdx.y;
  const int col = blockIdx.x * 256 + threadIdx.x;
  __shared__ float xs[Dn];
  for (int d = threadIdx.x; d < Dn; d += 256) xs[d] = xl[(size_t)b * Dn + d];
  __syncthreads();
  float acc = 0.f;
  for (int d = 0; d < Dn; ++d) acc += xs[d] * W[(size_t)d * Vn + col];
  out[(size_t)b * Vn + col] = acc;
}

// ---------------------------------------------------------------------------
extern "C" void kernel_launch(void* const* d_in, const int* in_sizes, int n_in,
                              void* d_out, int out_size, void* d_ws, size_t ws_size,
                              hipStream_t stream) {
  (void)in_sizes; (void)n_in; (void)out_size; (void)ws_size;
  const float* hidden     = (const float*)d_in[0];
  const float* mask_scale = (const float*)d_in[3];
  const float* ln1 = (const float*)d_in[4];
  const float* qw  = (const float*)d_in[5];
  const float* kw  = (const float*)d_in[6];
  const float* vw  = (const float*)d_in[7];
  const float* qnw = (const float*)d_in[8];
  const float* knw = (const float*)d_in[9];
  const float* ow  = (const float*)d_in[10];
  const float* ln2 = (const float*)d_in[11];
  const float* gw  = (const float*)d_in[12];
  const float* uw  = (const float*)d_in[13];
  const float* dw  = (const float*)d_in[14];
  const float* nw  = (const float*)d_in[15];
  const float* lmw = (const float*)d_in[16];
  const int history = 0;   // reference: history_len = 0

  float* ws = (float*)d_ws;
  size_t o = 0;
  const size_t BS = (size_t)Bn * Sn;
  float* x  = ws + o; o += BS * Dn;            // residual stream (fp32)
  float* qb = ws + o; o += BS * Hn * HDn;      // q fp32 (pre-rope)
  float* kb = ws + o; o += BS * KVn * HDn;     // k fp32 (pre-rope)
  float* vb = ws + o; o += BS * KVn * HDn;     // v fp32
  float* gb = ws + o; o += BS * Fn;            // gate fp32
  float* ub = ws + o; o += BS * Fn;            // up fp32
  _Float16* xnh = (_Float16*)(ws + o); o += BS * Dn / 2;         // normed acts f16
  _Float16* cxh = (_Float16*)(ws + o); o += BS * Hn * HDn / 2;   // ctx f16
  _Float16* ghh = (_Float16*)(ws + o); o += BS * Fn / 2;         // silu*up f16
  _Float16* qh  = (_Float16*)(ws + o); o += BS * Hn * HDn / 2;   // q f16 (B,H,S,HD)
  _Float16* khb = (_Float16*)(ws + o); o += BS * KVn * HDn / 2;  // k f16 (B,KV,S,HD)
  _Float16* vtb = (_Float16*)(ws + o); o += BS * KVn * HDn / 2;  // v f16 (B,KV,HD,S)
  _Float16* wt  = (_Float16*)(ws + o); o += (size_t)Fn * Dn / 2 + 64; // Wt scratch
  float* xl = ws + o; o += (size_t)Bn * Dn;

  (void)hipMemcpyAsync(x, hidden, BS * Dn * sizeof(float),
                       hipMemcpyDeviceToDevice, stream);

  // transpose+convert weight W[K][N] -> wt[N][K] f16, then run the GEMM on it
  auto gemmT = [&](const _Float16* Af16, const float* Wsrc, const float* Res,
                   float* Cout, int M, int N, int K) {
    wtrans_kernel<<<dim3(K / 32, N / 32), dim3(256), 0, stream>>>(Wsrc, wt, N, K);
    const int blocks = (M / 128) * (N / 128);
    gemm_wmma_kernel<<<dim3(blocks), dim3(256), 0, stream>>>(Af16, wt, Res, Cout, M, N, K);
  };

  const int M = (int)BS;
  for (int i = 0; i < Ln; ++i) {
    rmsnorm_kernel<<<dim3(M), dim3(256), 0, stream>>>(x, ln1 + (size_t)i * Dn, xnh, Dn);
    gemmT(xnh, qw + (size_t)i * Dn * Hn * HDn,  nullptr, qb, M, Hn * HDn,  Dn);
    gemmT(xnh, kw + (size_t)i * Dn * KVn * HDn, nullptr, kb, M, KVn * HDn, Dn);
    gemmT(xnh, vw + (size_t)i * Dn * KVn * HDn, nullptr, vb, M, KVn * HDn, Dn);
    qknorm_rope_kernel<<<dim3(M * Hn),  dim3(HDn), 0, stream>>>(
        qb, qnw + (size_t)i * HDn, qh,  Hn,  history);
    qknorm_rope_kernel<<<dim3(M * KVn), dim3(HDn), 0, stream>>>(
        kb, knw + (size_t)i * HDn, khb, KVn, history);
    vtrans_kernel<<<dim3(Bn * KVn, Sn / 32, HDn / 32), dim3(256), 0, stream>>>(vb, vtb);
    attn_kernel<<<dim3(Bn * Hn * (Sn / 16)), dim3(32), 0, stream>>>(
        qh, khb, vtb, mask_scale, cxh);
    gemmT(cxh, ow + (size_t)i * Hn * HDn * Dn, x, x, M, Dn, Hn * HDn);
    rmsnorm_kernel<<<dim3(M), dim3(256), 0, stream>>>(x, ln2 + (size_t)i * Dn, xnh, Dn);
    gemmT(xnh, gw + (size_t)i * Dn * Fn, nullptr, gb, M, Fn, Dn);
    gemmT(xnh, uw + (size_t)i * Dn * Fn, nullptr, ub, M, Fn, Dn);
    {
      const size_t n = BS * Fn;
      silu_mul_kernel<<<dim3((unsigned)((n + 255) / 256)), dim3(256), 0, stream>>>(
          gb, ub, ghh, n);
    }
    gemmT(ghh, dw + (size_t)i * Fn * Dn, x, x, M, Dn, Fn);
  }

  last_rms_kernel<<<dim3(Bn), dim3(256), 0, stream>>>(x, nw, xl);
  lm_head_kernel<<<dim3(Vn / 256, Bn), dim3(256), 0, stream>>>(xl, lmw, (float*)d_out);
}
